// VQ_split_70360154243692
// MI455X (gfx1250) — matrix operand order repaired and checked by reading, not compiled, not run on hardware
//
#include <hip/hip_runtime.h>

// ---------------------------------------------------------------------------
// Problem dimensions (from setup_inputs): B=8, S=2048, TD=4096, CD=1024
// ---------------------------------------------------------------------------
#define MROWS 16384   // B*S
#define TD    4096
#define CD    1024

typedef __attribute__((ext_vector_type(16))) __bf16 v16bf;
typedef __attribute__((ext_vector_type(8)))  __bf16 v8bf;
typedef __attribute__((ext_vector_type(4)))  __bf16 v4bf;
typedef __attribute__((ext_vector_type(8)))  float  v8f;
typedef __attribute__((ext_vector_type(4)))  unsigned v4u;
typedef __attribute__((ext_vector_type(8)))  int      v8i;
typedef __attribute__((ext_vector_type(4)))  int      v4i;

// GEMM tiling: block 128x128, BK=32, 8 waves (4x2), wave tile 32x64.
#define BM 128
#define BN 128
#define BK 32
#define LSTR 40                  // LDS row stride (bf16): 80B rows, 16B aligned, banks 20r%64 all-distinct
#define BUFE ((BM + BN) * LSTR)  // bf16 elems per double-buffer slot (A tile then B tile)

// TDM availability (this toolchain: 6-arg builtin per compile probe)
#if __has_builtin(__builtin_amdgcn_tensor_load_to_lds) && __has_builtin(__builtin_amdgcn_s_wait_tensorcnt)
#define HAVE_TDM 1
#endif

// ---------------------------------------------------------------------------
// Workspace layout (bytes)
// ---------------------------------------------------------------------------
#define WS_H     ((size_t)0)                       // h, f32, MROWS*CD        = 67108864 B
#define WS_FXQ   ((size_t)67108864)                // fxq, bf16, MROWS*CD     = 33554432 B
#define WS_WI    ((size_t)100663296)               // W_in bf16, CD*TD        =  8388608 B
#define WS_WO    ((size_t)109051904)               // W_out bf16, TD*CD       =  8388608 B
#define WS_PART  ((size_t)117440512)               // 2048*2 doubles          =    32768 B
#define WS_MS    ((size_t)117473280)               // mean/std, 2 floats (+pad)
#define WS_RL    ((size_t)117473296)               // rowLoss, 16384 floats   =    65536 B

#define STATS_BLOCKS 2048

// ---------------------------------------------------------------------------
// Weight conversion f32 -> bf16 (both weight matrices are 4,194,304 elements)
// ---------------------------------------------------------------------------
__global__ __launch_bounds__(256) void prep_kernel(const float* __restrict__ Wi_f,
                                                   const float* __restrict__ Wo_f,
                                                   __bf16* __restrict__ Wi_b,
                                                   __bf16* __restrict__ Wo_b) {
  size_t i = (size_t)blockIdx.x * blockDim.x + threadIdx.x;  // exactly CD*TD threads
  Wi_b[i] = (__bf16)Wi_f[i];
  Wo_b[i] = (__bf16)Wo_f[i];
}

// ---------------------------------------------------------------------------
// Helpers
// ---------------------------------------------------------------------------
static __device__ __forceinline__ v16bf make_frag(const __bf16* p0, const __bf16* p1) {
  v8bf lo = *(const v8bf*)p0;
  v8bf hi = *(const v8bf*)p1;
  return __builtin_shufflevector(lo, hi, 0,1,2,3,4,5,6,7,8,9,10,11,12,13,14,15);
}

#if defined(HAVE_TDM)
// Issue one TDM 2D tile load: tileRows x BK bf16 elements from a row-major
// [rows x K] bf16 matrix into LDS at byte offset lds_off, with LDS row padding
// of 4 DWORDs every 16 DWORDs -> reproduces the LSTR=40 layout in hardware.
static __device__ __forceinline__ void tdm_load_tile(const __bf16* gsrc, unsigned lds_off,
                                                     int K, int rowsTotal, int tileRows) {
  unsigned long long ga = (unsigned long long)(const void*)gsrc;
  // D# group 0: count=1 | lds_addr | global_addr[56:0] | type=2
  v4u g0 = { 1u,
             lds_off,
             (unsigned)ga,
             ((unsigned)((ga >> 32) & 0x01FFFFFFull)) | 0x80000000u };
  // D# group 1: data_size=2B, pad_enable, pad_interval=3 (16 DW), pad_amount=3 (4 DW)
  unsigned d0 = (1u << 16) | (1u << 20) | (3u << 22) | (3u << 25);
  unsigned uK = (unsigned)K, uM = (unsigned)rowsTotal;
  v8i g1 = { (int)d0,
             (int)((uK & 0xFFFFu) << 16),                    // tensor_dim0[15:0]
             (int)((uK >> 16) | ((uM & 0xFFFFu) << 16)),     // tensor_dim0[31:16] | tensor_dim1[15:0]
             (int)((uM >> 16) | ((unsigned)BK << 16)),       // tensor_dim1[31:16] | tile_dim0
             (int)tileRows,                                  // tile_dim1 | tile_dim2=0
             (int)uK,                                        // tensor_dim0_stride[31:0]
             0, 0 };                                         // stride0 hi, stride1 (unused, 2D)
  v4i z4 = { 0, 0, 0, 0 };
  v8i z8 = { 0, 0, 0, 0, 0, 0, 0, 0 };
  // 6-arg form on this toolchain: (g0, g1, g2, g3, g4, cpol)
  __builtin_amdgcn_tensor_load_to_lds(g0, g1, z4, z4, z8, 0);
}
#endif

// ---------------------------------------------------------------------------
// TN GEMM: C[m,n] = sum_k A[m,k]*B[n,k] + bias[n]
//   A: [M,K] row-major, f32 (converted in-flight, register double-buffer) or
//      bf16 (staged by the Tensor Data Mover, async double-buffer).
//   B: [N,K] row-major bf16.
// ---------------------------------------------------------------------------
template <bool A_IS_F32>
__global__ __launch_bounds__(256) void gemm_bf16_tn(const void* __restrict__ Aptr,
                                                    const __bf16* __restrict__ Bmat,
                                                    const float* __restrict__ bias,
                                                    float* __restrict__ C,
                                                    int M, int N, int K) {
  __shared__ __bf16 smem[2 * BUFE];   // 40,960 B total; static LDS base = offset 0

  const int tid  = threadIdx.x;
  const int nBlk = N / BN;
  const int bm   = blockIdx.x / nBlk;   // N-fast ordering: A strip reused through L2
  const int bn   = blockIdx.x % nBlk;

  const int wave = tid >> 5;            // 0..7
  const int lane = tid & 31;
  const int wm   = wave >> 1;           // 0..3 -> 32-row strip
  const int wn   = wave & 1;            // 0..1 -> 64-col strip
  const int lr   = lane & 15;
  const int lh   = lane >> 4;           // 0/1

  v8f acc[2][4];
  const v8f vzero = {0.f, 0.f, 0.f, 0.f, 0.f, 0.f, 0.f, 0.f};
#pragma unroll
  for (int t = 0; t < 2; ++t)
#pragma unroll
    for (int u = 0; u < 4; ++u) acc[t][u] = vzero;

  const int nt = K / BK;

#if defined(HAVE_TDM)
  constexpr bool useTdm = !A_IS_F32;
#else
  constexpr bool useTdm = false;
#endif

  if (useTdm) {
#if defined(HAVE_TDM)
    // -------- async double-buffered staging via Tensor Data Mover --------
    const __bf16* Ab = (const __bf16*)Aptr;
    if (wave == 0) {
      tdm_load_tile(Ab + (size_t)bm * BM * K, 0u, K, M, BM);
      tdm_load_tile(Bmat + (size_t)bn * BN * K, (unsigned)(BM * LSTR * 2), K, N, BN);
    }
    for (int i = 0; i < nt; ++i) {
      if (wave == 0) {
        if (i + 1 < nt) {
          const unsigned base = (unsigned)(((i + 1) & 1) * BUFE * 2);
          tdm_load_tile(Ab + (size_t)bm * BM * K + (i + 1) * BK, base, K, M, BM);
          tdm_load_tile(Bmat + (size_t)bn * BN * K + (i + 1) * BK,
                        base + (unsigned)(BM * LSTR * 2), K, N, BN);
          __builtin_amdgcn_s_wait_tensorcnt((short)2);  // tile i landed; i+1 in flight
        } else {
          __builtin_amdgcn_s_wait_tensorcnt((short)0);
        }
      }
      __syncthreads();                                   // publish buffer i
      {
        const __bf16* As = &smem[(i & 1) * BUFE];
        const __bf16* Bs = As + BM * LSTR;
        v16bf af[2], bfv[4];
#pragma unroll
        for (int t = 0; t < 2; ++t) {
          const __bf16* p = &As[(wm * 32 + t * 16 + lr) * LSTR + lh * 8];
          af[t] = make_frag(p, p + 16);
        }
#pragma unroll
        for (int u = 0; u < 4; ++u) {
          const __bf16* p = &Bs[(wn * 64 + u * 16 + lr) * LSTR + lh * 8];
          bfv[u] = make_frag(p, p + 16);
        }
#pragma unroll
        for (int t = 0; t < 2; ++t)
#pragma unroll
          for (int u = 0; u < 4; ++u)
            acc[t][u] = __builtin_amdgcn_wmma_f32_16x16x32_bf16(
                false, af[t], false, bfv[u], (short)0, acc[t][u], false, false);
      }
      __syncthreads();                                   // reads done before TDM reuses buffer
    }
#endif
  } else {
    // -------- register double-buffered staging (f32 A converted in flight) --------
    const int ldRow  = tid >> 1;        // 0..127
    const int ldHalf = tid & 1;         // each thread covers 16 k-elems
    float4 ra[4];
    v8bf   rab[2];
    v8bf   rb[2];

    auto load_regs = [&](int k0) {
      if (A_IS_F32) {
        const float* A = (const float*)Aptr;
        const float4* s = (const float4*)(A + (size_t)(bm * BM + ldRow) * K + k0 + ldHalf * 16);
#pragma unroll
        for (int r = 0; r < 4; ++r) ra[r] = s[r];
      } else {
        const __bf16* A = (const __bf16*)Aptr;
        const v8bf* s = (const v8bf*)(A + (size_t)(bm * BM + ldRow) * K + k0 + ldHalf * 16);
        rab[0] = s[0]; rab[1] = s[1];
      }
      const v8bf* s = (const v8bf*)(Bmat + (size_t)(bn * BN + ldRow) * K + k0 + ldHalf * 16);
      rb[0] = s[0]; rb[1] = s[1];
    };
    auto store_regs = [&](int buf) {
      __bf16* dstA = &smem[buf * BUFE + ldRow * LSTR + ldHalf * 16];
      if (A_IS_F32) {
#pragma unroll
        for (int r = 0; r < 4; ++r) {
          v4bf w = {(__bf16)ra[r].x, (__bf16)ra[r].y, (__bf16)ra[r].z, (__bf16)ra[r].w};
          *(v4bf*)(dstA + r * 4) = w;
        }
      } else {
        *(v8bf*)dstA = rab[0]; *(v8bf*)(dstA + 8) = rab[1];
      }
      __bf16* dstB = dstA + BM * LSTR;
      *(v8bf*)dstB = rb[0]; *(v8bf*)(dstB + 8) = rb[1];
    };

    load_regs(0);
    store_regs(0);
    for (int i = 0; i < nt; ++i) {
      __syncthreads();                       // buffer i visible to all waves
      if (i + 1 < nt) load_regs((i + 1) * BK);   // global loads fly during compute
      {
        const __bf16* As = &smem[(i & 1) * BUFE];
        const __bf16* Bs = As + BM * LSTR;
        v16bf af[2], bfv[4];
#pragma unroll
        for (int t = 0; t < 2; ++t) {
          const __bf16* p = &As[(wm * 32 + t * 16 + lr) * LSTR + lh * 8];
          af[t] = make_frag(p, p + 16);
        }
#pragma unroll
        for (int u = 0; u < 4; ++u) {
          const __bf16* p = &Bs[(wn * 64 + u * 16 + lr) * LSTR + lh * 8];
          bfv[u] = make_frag(p, p + 16);
        }
#pragma unroll
        for (int t = 0; t < 2; ++t)
#pragma unroll
          for (int u = 0; u < 4; ++u)
            acc[t][u] = __builtin_amdgcn_wmma_f32_16x16x32_bf16(
                false, af[t], false, bfv[u], (short)0, acc[t][u], false, false);
      }
      __syncthreads();                       // reads of buffer i complete
      if (i + 1 < nt) store_regs((i + 1) & 1);
    }
  }

  // ---- epilogue: lane<16: VGPR j holds M=j; lane>=16: M=j+8 ----
#pragma unroll
  for (int t = 0; t < 2; ++t) {
    const int rbase = bm * BM + wm * 32 + t * 16 + lh * 8;
#pragma unroll
    for (int u = 0; u < 4; ++u) {
      const int col = bn * BN + wn * 64 + u * 16 + lr;
      const float bv = bias[col];
      float* p = C + (size_t)rbase * N + col;
#pragma unroll
      for (int j = 0; j < 8; ++j) { *p = acc[t][u][j] + bv; p += N; }
    }
  }
}

// ---------------------------------------------------------------------------
// Global stats over h: deterministic two-pass (per-block f64 partials)
// ---------------------------------------------------------------------------
__global__ __launch_bounds__(256) void stats_partial(const float* __restrict__ h,
                                                     double* __restrict__ part, size_t n) {
  __shared__ double ss[256], ss2[256];
  double s = 0.0, s2 = 0.0;
  for (size_t i = (size_t)blockIdx.x * blockDim.x + threadIdx.x; i < n;
       i += (size_t)gridDim.x * blockDim.x) {
    double v = (double)h[i];
    s += v; s2 += v * v;
  }
  ss[threadIdx.x] = s; ss2[threadIdx.x] = s2;
  __syncthreads();
  for (int off = 128; off > 0; off >>= 1) {
    if (threadIdx.x < off) { ss[threadIdx.x] += ss[threadIdx.x + off];
                             ss2[threadIdx.x] += ss2[threadIdx.x + off]; }
    __syncthreads();
  }
  if (threadIdx.x == 0) { part[blockIdx.x] = ss[0]; part[STATS_BLOCKS + blockIdx.x] = ss2[0]; }
}

__global__ __launch_bounds__(256) void stats_final(const double* __restrict__ part,
                                                   float* __restrict__ meanstd) {
  __shared__ double ss[256], ss2[256];
  double s = 0.0, s2 = 0.0;
  for (int i = threadIdx.x; i < STATS_BLOCKS; i += 256) {
    s += part[i]; s2 += part[STATS_BLOCKS + i];
  }
  ss[threadIdx.x] = s; ss2[threadIdx.x] = s2;
  __syncthreads();
  for (int off = 128; off > 0; off >>= 1) {
    if (threadIdx.x < off) { ss[threadIdx.x] += ss[threadIdx.x + off];
                             ss2[threadIdx.x] += ss2[threadIdx.x + off]; }
    __syncthreads();
  }
  if (threadIdx.x == 0) {
    const double n = (double)MROWS * (double)CD;
    double mean = ss[0] / n;
    double var  = (ss2[0] - n * mean * mean) / (n - 1.0);   // ddof=1
    meanstd[0] = (float)mean;
    meanstd[1] = (float)sqrt(var);
  }
}

// ---------------------------------------------------------------------------
// Quantize: clip/normalize, 2-bit round (pack/unpack is identity), write bf16
// fxq, per-row commitment loss partial. One block per row (1024 cols).
// ---------------------------------------------------------------------------
__global__ __launch_bounds__(256) void quantize_kernel(const float* __restrict__ h,
                                                       const float* __restrict__ meanstd,
                                                       __bf16* __restrict__ fxq,
                                                       float* __restrict__ rowLoss) {
  const int row = blockIdx.x;
  const int tid = threadIdx.x;
  const float mean = meanstd[0], sd = meanstd[1];
  const float qmin = mean - 3.0f * sd;
  const float qmax = mean + 3.0f * sd;
  const float denom = qmax - qmin + 1e-4f;

  const float4 hv = *(const float4*)(h + (size_t)row * CD + tid * 4);
  const float vv[4] = {hv.x, hv.y, hv.z, hv.w};

  float d = 0.f, n1 = 0.f, n2 = 0.f;
#pragma unroll
  for (int i = 0; i < 4; ++i) {
    float hc = fminf(fmaxf(vv[i], qmin), qmax);
    float fx = 2.0f * (hc - qmin + 1e-4f) / denom - 1.0f;        // in [-1,1]
    float xr = rintf(fx * 1.5f - 0.5f);                          // round-half-even
    float q  = (xr + 0.5f) * (1.0f / 1.5f);                      // {-1,-1/3,1/3,1}
    d  += fx * q;  n1 += fx * fx;  n2 += q * q;
    fxq[(size_t)row * CD + tid * 4 + i] = (__bf16)q;
  }

  __shared__ float sd_[256], sn1[256], sn2[256];
  sd_[tid] = d; sn1[tid] = n1; sn2[tid] = n2;
  __syncthreads();
  for (int off = 128; off > 0; off >>= 1) {
    if (tid < off) { sd_[tid] += sd_[tid + off]; sn1[tid] += sn1[tid + off];
                     sn2[tid] += sn2[tid + off]; }
    __syncthreads();
  }
  if (tid == 0) {
    float sim = sd_[0] / (sqrtf(sn1[0]) * sqrtf(sn2[0]) + 1e-3f);
    rowLoss[row] = 1.0f - sim;
  }
}

// ---------------------------------------------------------------------------
// Deterministic reduction of the 16384 per-row losses -> scalar L_comm
// ---------------------------------------------------------------------------
__global__ __launch_bounds__(256) void loss_final(const float* __restrict__ rowLoss,
                                                  float* __restrict__ out_scalar) {
  __shared__ double ss[256];
  double s = 0.0;
  for (int i = threadIdx.x; i < MROWS; i += 256) s += (double)rowLoss[i];
  ss[threadIdx.x] = s;
  __syncthreads();
  for (int off = 128; off > 0; off >>= 1) {
    if (threadIdx.x < off) ss[threadIdx.x] += ss[threadIdx.x + off];
    __syncthreads();
  }
  if (threadIdx.x == 0) out_scalar[0] = (float)(ss[0] / (double)MROWS);
}

// ---------------------------------------------------------------------------
// Launcher
// ---------------------------------------------------------------------------
extern "C" void kernel_launch(void* const* d_in, const int* in_sizes, int n_in,
                              void* d_out, int out_size, void* d_ws, size_t ws_size,
                              hipStream_t stream) {
  const float* x     = (const float*)d_in[0];   // [8,2048,4096]
  const float* W_in  = (const float*)d_in[1];   // [1024,4096]
  const float* b_in  = (const float*)d_in[2];   // [1024]
  const float* W_out = (const float*)d_in[3];   // [4096,1024]
  const float* b_out = (const float*)d_in[4];   // [4096]

  char* ws = (char*)d_ws;
  float*  h       = (float*)(ws + WS_H);
  __bf16* fxq     = (__bf16*)(ws + WS_FXQ);
  __bf16* Wi_b    = (__bf16*)(ws + WS_WI);
  __bf16* Wo_b    = (__bf16*)(ws + WS_WO);
  double* part    = (double*)(ws + WS_PART);
  float*  meanstd = (float*)(ws + WS_MS);
  float*  rowLoss = (float*)(ws + WS_RL);

  float* out = (float*)d_out;                        // [16384, 4096]
  float* L_comm = out + (size_t)MROWS * TD;          // trailing scalar

  // 1) weights -> bf16 (CD*TD = 4,194,304 elements each)
  prep_kernel<<<(CD * TD) / 256, 256, 0, stream>>>(W_in, W_out, Wi_b, Wo_b);

  // 2) h = x @ W_in^T + b_in   (M=16384, N=1024, K=4096)
  gemm_bf16_tn<true><<<(MROWS / BM) * (CD / BN), 256, 0, stream>>>(
      (const void*)x, Wi_b, b_in, h, MROWS, CD, TD);

  // 3) global mean/std (deterministic two-pass)
  stats_partial<<<STATS_BLOCKS, 256, 0, stream>>>(h, part, (size_t)MROWS * CD);
  stats_final<<<1, 256, 0, stream>>>(part, meanstd);

  // 4) quantize + per-row commitment loss
  quantize_kernel<<<MROWS, 256, 0, stream>>>(h, meanstd, fxq, rowLoss);

  // 5) out = fxq @ W_out^T + b_out   (M=16384, N=4096, K=1024)  — TDM-staged
  gemm_bf16_tn<false><<<(MROWS / BM) * (TD / BN), 256, 0, stream>>>(
      (const void*)fxq, Wo_b, b_out, out, MROWS, TD, CD);

  // 6) L_comm scalar
  loss_final<<<1, 256, 0, stream>>>(rowLoss, L_comm);
}
